// MultiHeadAttention_3298534884057
// MI455X (gfx1250) — matrix operand order repaired and checked by reading, not compiled, not run on hardware
//
#include <hip/hip_runtime.h>
#include <hip/hip_bf16.h>

#define BB 4
#define SS 2048
#define DD 1024
#define HH 16
// head dims Q = V = 64

typedef _Float16 half_t;
typedef __attribute__((ext_vector_type(16))) _Float16 v16h;
typedef __attribute__((ext_vector_type(8)))  _Float16 v8h;
typedef __attribute__((ext_vector_type(8)))  float    v8f;

static __device__ __forceinline__ v8f wmma_f16f32(v16h a, v16h b, v8f c) {
  // D = A(16x32 f16) * B(32x16 f16) + C(16x16 f32)
  return __builtin_amdgcn_wmma_f32_16x16x32_f16(false, a, false, b, (short)0, c,
                                                false, false);
}

// A fragment, 16x32 f16, row-major source with leading dim ld (halves).
// lane l: row = l&15, K chunks [8*(l>>4) .. +7] and [+16 .. +23].
static __device__ __forceinline__ v16h load_a_f16(const half_t* __restrict__ src,
                                                  int ld, int lane) {
  const half_t* p = src + (size_t)(lane & 15) * ld + ((lane >> 4) << 3);
  v8h lo = *(const v8h*)p;
  v8h hi = *(const v8h*)(p + 16);
  v16h r;
#pragma unroll
  for (int i = 0; i < 8; ++i) { r[i] = lo[i]; r[i + 8] = hi[i]; }
  return r;
}

static __device__ __forceinline__ v16h load_a_f32(const float* __restrict__ src,
                                                  int ld, int lane) {
  const float* p = src + (size_t)(lane & 15) * ld + ((lane >> 4) << 3);
  v16h r;
#pragma unroll
  for (int i = 0; i < 8; ++i) r[i] = (_Float16)p[i];
#pragma unroll
  for (int i = 0; i < 8; ++i) r[i + 8] = (_Float16)p[16 + i];
  return r;
}

// B fragment, 32x16 f16, row-major source: lane l holds col n=l&15, K rows [8*(l>>4) .. +7] and [+16 .. +23].
static __device__ __forceinline__ v16h load_b_f16(const half_t* __restrict__ src,
                                                  int ld, int lane) {
  const half_t* p = src + (size_t)((lane >> 4) << 3) * ld + (lane & 15);
  v16h r;
#pragma unroll
  for (int i = 0; i < 8; ++i) { r[i] = p[(size_t)i * ld]; r[i + 8] = p[(size_t)(16 + i) * ld]; }
  return r;
}

// ---------------------------------------------------------------------------
// fp32 -> f16 conversion (weights only; grid-stride)
// ---------------------------------------------------------------------------
__global__ __launch_bounds__(256) void cvt_f32_f16_kernel(
    const float* __restrict__ src, half_t* __restrict__ dst, int n) {
  int i = blockIdx.x * 256 + threadIdx.x;
  int stride = gridDim.x * 256;
  for (; i < n; i += stride) dst[i] = (half_t)src[i];
}

// ---------------------------------------------------------------------------
// QKV projection: out[b,h] = (X[b] (SSxDD) @ W16[h] (DDx64) + bias[h])*outscale
// grid = (SS/64, BB*HH), block = 128: wave wv owns rows [m0, m0+16) x all 64
// cols; one A fragment feeds 4 WMMAs per K-step.
// transpose==1 stores out as [bh, 64, SS] (for K), else [bh, SS, 64].
// ---------------------------------------------------------------------------
__global__ __launch_bounds__(128) void mha_proj_kernel(
    const float* __restrict__ X, const half_t* __restrict__ W16,
    const float* __restrict__ bias, half_t* __restrict__ out,
    int transpose, float outscale) {
  const int lane = threadIdx.x & 31;
  const int wv = threadIdx.x >> 5;
  const int m0 = blockIdx.x * 64 + wv * 16;
  const int bh = blockIdx.y;
  const int b = bh >> 4, h = bh & 15;  // HH == 16

  const float* Xb = X + (size_t)b * SS * DD + (size_t)m0 * DD;
  const half_t* Wh = W16 + (size_t)h * DD * 64;

  v8f acc[4] = {};
  for (int k0 = 0; k0 < DD; k0 += 32) {
    if (k0 + 32 < DD)
      __builtin_prefetch(Xb + k0 + 32 + (size_t)(lane & 15) * DD, 0, 1);
    v16h a = load_a_f32(Xb + k0, DD, lane);
#pragma unroll
    for (int nt = 0; nt < 4; ++nt)
      acc[nt] = wmma_f16f32(a, load_b_f16(Wh + (size_t)k0 * 64 + nt * 16, 64, lane),
                            acc[nt]);
  }

  const int col = lane & 15;
  const int rowbase = (lane >> 4) << 3;
  if (!transpose) {
    half_t* po = out + ((size_t)bh * SS + m0 + rowbase) * 64 + col;
#pragma unroll
    for (int nt = 0; nt < 4; ++nt) {
      float bval = bias[h * 64 + nt * 16 + col];
#pragma unroll
      for (int r = 0; r < 8; ++r)
        po[(size_t)r * 64 + nt * 16] = (half_t)((acc[nt][r] + bval) * outscale);
    }
  } else {
    half_t* po = out + ((size_t)bh * 64 + col) * SS + m0 + rowbase;
#pragma unroll
    for (int nt = 0; nt < 4; ++nt) {
      float bval = bias[h * 64 + nt * 16 + col];
#pragma unroll
      for (int r = 0; r < 8; ++r)
        po[(size_t)nt * 16 * SS + r] = (half_t)((acc[nt][r] + bval) * outscale);
    }
  }
}

// ---------------------------------------------------------------------------
// Flash attention: one wave per 16 query rows per (b,h).
// grid = (SS/16, BB*HH), block = 32.
// Qp [bh,SS,64] f16 (pre-scaled by 1/sqrt(S)), KpT [bh,64,SS], Vp [bh,SS,64].
// Writes attn-cat [BB,SS,HH*64] f16.
// ---------------------------------------------------------------------------
__global__ __launch_bounds__(32) void mha_flash_kernel(
    const half_t* __restrict__ Qp, const half_t* __restrict__ KpT,
    const half_t* __restrict__ Vp, half_t* __restrict__ attn) {
  const int lane = threadIdx.x & 31;
  const int m0 = blockIdx.x * 16;
  const int bh = blockIdx.y;
  const int b = bh >> 4, h = bh & 15;

  const half_t* Qb = Qp + (size_t)bh * SS * 64;
  const half_t* KTb = KpT + (size_t)bh * 64 * SS;
  const half_t* Vb = Vp + (size_t)bh * SS * 64;

  const v16h aQ0 = load_a_f16(Qb + (size_t)m0 * 64, 64, lane);
  const v16h aQ1 = load_a_f16(Qb + (size_t)m0 * 64 + 32, 64, lane);

  float mrow[8], lrow[8];
#pragma unroll
  for (int r = 0; r < 8; ++r) { mrow[r] = -__builtin_inff(); lrow[r] = 0.0f; }
  v8f O0 = {}, O1 = {}, O2 = {}, O3 = {};

  __shared__ __align__(16) half_t Plds[16 * 32];
  const int col = lane & 15;
  const int rowbase = (lane >> 4) << 3;

  for (int t0 = 0; t0 < SS; t0 += 32) {
    // score tiles: [16 q] x [32 keys], K-dim = 64 in two WMMA steps
    v8f s0 = {}, s1 = {};
    s0 = wmma_f16f32(aQ0, load_b_f16(KTb + t0, SS, lane), s0);
    s0 = wmma_f16f32(aQ1, load_b_f16(KTb + (size_t)32 * SS + t0, SS, lane), s0);
    s1 = wmma_f16f32(aQ0, load_b_f16(KTb + t0 + 16, SS, lane), s1);
    s1 = wmma_f16f32(aQ1, load_b_f16(KTb + (size_t)32 * SS + t0 + 16, SS, lane), s1);

    float p0[8], p1[8], alpha[8];
#pragma unroll
    for (int r = 0; r < 8; ++r) {
      float a0 = s0[r];
      float a1 = s1[r];
      float vmax = fmaxf(a0, a1);
      vmax = fmaxf(vmax, __shfl_xor(vmax, 1, 16));
      vmax = fmaxf(vmax, __shfl_xor(vmax, 2, 16));
      vmax = fmaxf(vmax, __shfl_xor(vmax, 4, 16));
      vmax = fmaxf(vmax, __shfl_xor(vmax, 8, 16));
      float mnew = fmaxf(mrow[r], vmax);
      alpha[r] = __expf(mrow[r] - mnew);
      mrow[r] = mnew;
      p0[r] = __expf(a0 - mnew);
      p1[r] = __expf(a1 - mnew);
      float rs = p0[r] + p1[r];
      rs += __shfl_xor(rs, 1, 16);
      rs += __shfl_xor(rs, 2, 16);
      rs += __shfl_xor(rs, 4, 16);
      rs += __shfl_xor(rs, 8, 16);
      lrow[r] = lrow[r] * alpha[r] + rs;
    }
#pragma unroll
    for (int r = 0; r < 8; ++r) {
      O0[r] *= alpha[r]; O1[r] *= alpha[r];
      O2[r] *= alpha[r]; O3[r] *= alpha[r];
    }
    // Stage P (16x32 f16) via LDS: C-layout -> A-layout conversion.
#pragma unroll
    for (int r = 0; r < 8; ++r) {
      Plds[(rowbase + r) * 32 + col]      = (half_t)p0[r];
      Plds[(rowbase + r) * 32 + col + 16] = (half_t)p1[r];
    }
    __syncthreads();
    v16h aP = load_a_f16(Plds, 32, lane);
    O0 = wmma_f16f32(aP, load_b_f16(Vb + (size_t)t0 * 64, 64, lane), O0);
    O1 = wmma_f16f32(aP, load_b_f16(Vb + (size_t)t0 * 64 + 16, 64, lane), O1);
    O2 = wmma_f16f32(aP, load_b_f16(Vb + (size_t)t0 * 64 + 32, 64, lane), O2);
    O3 = wmma_f16f32(aP, load_b_f16(Vb + (size_t)t0 * 64 + 48, 64, lane), O3);
    __syncthreads();
  }

  float inv[8];
#pragma unroll
  for (int r = 0; r < 8; ++r) inv[r] = 1.0f / lrow[r];

  half_t* outp = attn + ((size_t)b * SS + m0) * (HH * 64) + h * 64;
#pragma unroll
  for (int r = 0; r < 8; ++r) {
    size_t row = (size_t)(rowbase + r) * (HH * 64);
    outp[row + col]      = (half_t)(O0[r] * inv[r]);
    outp[row + 16 + col] = (half_t)(O1[r] * inv[r]);
    outp[row + 32 + col] = (half_t)(O2[r] * inv[r]);
    outp[row + 48 + col] = (half_t)(O3[r] * inv[r]);
  }
}

// ---------------------------------------------------------------------------
// Output projection: attn-cat [B*S,1024] f16 @ Wo16 [1024,1024] f16 + bo -> f32
// grid = (B*S/64, 1024/64), block = 128: wave owns 16x64 tile, A reused 4x.
// ---------------------------------------------------------------------------
__global__ __launch_bounds__(128) void mha_outproj_kernel(
    const half_t* __restrict__ A, const half_t* __restrict__ Wo16,
    const float* __restrict__ bo, float* __restrict__ out) {
  const int lane = threadIdx.x & 31;
  const int wv = threadIdx.x >> 5;
  const int m0 = blockIdx.x * 64 + wv * 16;
  const int n0 = blockIdx.y * 64;

  v8f acc[4] = {};
  for (int k0 = 0; k0 < HH * 64; k0 += 32) {
    if (k0 + 32 < HH * 64)
      __builtin_prefetch(A + (size_t)(m0 + (lane & 15)) * (HH * 64) + k0 + 32, 0, 1);
    v16h a = load_a_f16(A + (size_t)m0 * (HH * 64) + k0, HH * 64, lane);
#pragma unroll
    for (int nt = 0; nt < 4; ++nt)
      acc[nt] = wmma_f16f32(
          a, load_b_f16(Wo16 + (size_t)k0 * DD + n0 + nt * 16, DD, lane), acc[nt]);
  }

  const int col = lane & 15;
  const int rowbase = (lane >> 4) << 3;
  float* po = out + (size_t)(m0 + rowbase) * DD + n0 + col;
#pragma unroll
  for (int nt = 0; nt < 4; ++nt) {
    float bval = bo[n0 + nt * 16 + col];
#pragma unroll
    for (int r = 0; r < 8; ++r)
      po[(size_t)r * DD + nt * 16] = acc[nt][r] + bval;
  }
}

extern "C" void kernel_launch(void* const* d_in, const int* in_sizes, int n_in,
                              void* d_out, int out_size, void* d_ws, size_t ws_size,
                              hipStream_t stream) {
  (void)in_sizes; (void)n_in; (void)out_size; (void)ws_size;
  const float* query = (const float*)d_in[0];
  const float* key   = (const float*)d_in[1];
  const float* value = (const float*)d_in[2];
  const float* Wq = (const float*)d_in[3];
  const float* bq = (const float*)d_in[4];
  const float* Wk = (const float*)d_in[5];
  const float* bk = (const float*)d_in[6];
  const float* Wv = (const float*)d_in[7];
  const float* bv = (const float*)d_in[8];
  const float* Wo = (const float*)d_in[9];
  const float* bo = (const float*)d_in[10];

  const size_t SZ = (size_t)BB * HH * SS * 64 * sizeof(half_t);  // 16 MiB each
  const size_t WZ = (size_t)HH * DD * 64 * sizeof(half_t);       // 2 MiB each
  char* ws = (char*)d_ws;
  half_t* Qp   = (half_t*)(ws);
  half_t* KpT  = (half_t*)(ws + SZ);
  half_t* Vp   = (half_t*)(ws + 2 * SZ);
  half_t* attn = (half_t*)(ws + 3 * SZ);
  half_t* Wq16 = (half_t*)(ws + 4 * SZ);
  half_t* Wk16 = (half_t*)(ws + 4 * SZ + WZ);
  half_t* Wv16 = (half_t*)(ws + 4 * SZ + 2 * WZ);
  half_t* Wo16 = (half_t*)(ws + 4 * SZ + 3 * WZ);

  const int wn = HH * DD * 64;  // == DD*DD
  cvt_f32_f16_kernel<<<1024, 256, 0, stream>>>(Wq, Wq16, wn);
  cvt_f32_f16_kernel<<<1024, 256, 0, stream>>>(Wk, Wk16, wn);
  cvt_f32_f16_kernel<<<1024, 256, 0, stream>>>(Wv, Wv16, wn);
  cvt_f32_f16_kernel<<<1024, 256, 0, stream>>>(Wo, Wo16, DD * DD);

  const float qscale = 0.02209708691207961f;  // 1/sqrt(S), per reference
  dim3 gp(SS / 64, BB * HH);
  mha_proj_kernel<<<gp, 128, 0, stream>>>(query, Wq16, bq, Qp, 0, qscale);
  mha_proj_kernel<<<gp, 128, 0, stream>>>(key,   Wk16, bk, KpT, 1, 1.0f);
  mha_proj_kernel<<<gp, 128, 0, stream>>>(value, Wv16, bv, Vp, 0, 1.0f);
  mha_flash_kernel<<<dim3(SS / 16, BB * HH), 32, 0, stream>>>(Qp, KpT, Vp, attn);
  mha_outproj_kernel<<<dim3(BB * SS / 64, DD / 64), 128, 0, stream>>>(
      attn, Wo16, bo, (float*)d_out);
}
